// Transformer_78975858639437
// MI455X (gfx1250) — matrix operand order repaired and checked by reading, not compiled, not run on hardware
//
#include <hip/hip_runtime.h>
#include <hip/hip_bf16.h>

// MI455X / gfx1250, wave32. One workgroup per batch (B=64), 512 threads = 16 waves.
// Output slice [:,0,:] lets us contract the q-side first: `keys` (16.8 MB total)
// is the only big HBM read. It is DMA'd global->LDS (f32, async-to-LDS via the
// gfx1250 GLOBAL_LOAD_ASYNC_TO_LDS_B128 path), scores AND the attn@keys
// contraction both run on v_wmma_f32_16x16x32_f16.

typedef _Float16 v16h __attribute__((ext_vector_type(16)));
typedef _Float16 v8h  __attribute__((ext_vector_type(8)));
typedef _Float16 v4h  __attribute__((ext_vector_type(4)));
typedef float    v8f  __attribute__((ext_vector_type(8)));
typedef float    v4f  __attribute__((ext_vector_type(4)));
typedef int      v4i  __attribute__((ext_vector_type(4)));

#define TSEQ 1024
#define DMODEL 64
#define NHEAD 4
#define NTHREADS 512
#define NEGF (-4294967295.0f)   // -2^32 + 1

// ---- LDS byte offsets (all >=32B aligned); total 320384 <= 320KB/WG ----
#define OFF_KEYS    0         // 1024*64 f32 = 262144
#define OFF_SCORES  262144    // 1024*4  f32 =  16384 (dead after softmax)
#define OFF_CPART   262144    //  4*4*256 f32 = 16384 (overlays scores)
#define OFF_PT      278528    // 16*1024 f16 =  32768
#define OFF_KM      311296    // 1024    i32 =   4096
#define OFF_UB      315392    // 16*64   f16 =   2048
#define OFF_Q0      317440    // 64      f32 =    256
#define OFF_QH      317696    // 64      f32 =    256
#define OFF_CTX     317952    // 4*64    f32 =   1024
#define OFF_RES     318976    // 64      f32 =    256
#define OFF_HID     319232    // 256     f32 =   1024
#define OFF_LSUM    320256    // 4       f32 (+pad)
#define SMEM_BYTES  320384

#if __has_builtin(__builtin_amdgcn_global_load_async_to_lds_b128) && \
    __has_builtin(__builtin_amdgcn_s_wait_asynccnt)
#define USE_ASYNC_LDS 1
#endif

union V16U { v16h v; v8h h[2]; };
union V8HU { v8h  v; v4h q[2]; };

#ifdef USE_ASYNC_LDS
__device__ __forceinline__ void async_copy16(const void* g, void* l) {
    // Builtin signature (from hipcc diagnostic): (v4i AS1*, v4i AS3*, imm, imm)
    __builtin_amdgcn_global_load_async_to_lds_b128(
        (__attribute__((address_space(1))) v4i*)g,
        (__attribute__((address_space(3))) v4i*)l, /*offset=*/0, /*cpol=*/0);
}
#endif

__global__ __launch_bounds__(NTHREADS)
void attn_block_kernel(const float* __restrict__ queries,
                       const float* __restrict__ keys,
                       const int*   __restrict__ query_mask,
                       const int*   __restrict__ key_mask,
                       const float* __restrict__ W_Q,
                       const float* __restrict__ W_K,
                       const float* __restrict__ W_V,
                       const float* __restrict__ fw1,
                       const float* __restrict__ fw2,
                       const float* __restrict__ fb1,
                       const float* __restrict__ fb2,
                       float*       __restrict__ out)
{
    extern __shared__ char smem[];
    float*    s_keys   = (float*)   (smem + OFF_KEYS);     // [1024][64] f32
    float*    s_scores = (float*)   (smem + OFF_SCORES);   // [1024][4]
    float*    s_cpart  = (float*)   (smem + OFF_CPART);    // [4][4][256]
    _Float16* s_pT     = (_Float16*)(smem + OFF_PT);       // [16][1024]
    int*      s_km     = (int*)     (smem + OFF_KM);
    _Float16* s_uB     = (_Float16*)(smem + OFF_UB);       // [16][64]
    float*    s_q0     = (float*)   (smem + OFF_Q0);
    float*    s_qh     = (float*)   (smem + OFF_QH);
    float*    s_ctx    = (float*)   (smem + OFF_CTX);
    float*    s_res    = (float*)   (smem + OFF_RES);
    float*    s_hid    = (float*)   (smem + OFF_HID);
    float*    s_lsum   = (float*)   (smem + OFF_LSUM);

    const int tid  = threadIdx.x;
    const int lane = tid & 31;
    const int wave = tid >> 5;
    const int b    = blockIdx.x;

    const float* kg = keys + (size_t)b * TSEQ * DMODEL;

    // ---------------- Phase 0: stage keys[b] -> LDS f32 (async DMA) --------
#ifdef USE_ASYNC_LDS
    for (int i = tid; i < TSEQ * DMODEL / 4; i += NTHREADS)
        async_copy16((const v4f*)kg + i, (v4f*)s_keys + i);
#else
    for (int i = tid; i < TSEQ * DMODEL / 4; i += NTHREADS) {
        __builtin_prefetch((const v4f*)kg + i + 2 * NTHREADS, 0, 0); // global_prefetch_b8
        ((v4f*)s_keys)[i] = ((const v4f*)kg)[i];
    }
#endif
    for (int i = tid; i < TSEQ; i += NTHREADS) s_km[i] = key_mask[b * TSEQ + i];
    for (int i = tid; i < DMODEL; i += NTHREADS)
        s_q0[i] = queries[(size_t)b * TSEQ * DMODEL + i];          // query row 0 only
    for (int i = tid; i < 16 * DMODEL; i += NTHREADS) s_uB[i] = (_Float16)0.0f;
    for (int i = tid; i < 12 * TSEQ; i += NTHREADS)                // zero-pad pT heads 4..15
        s_pT[NHEAD * TSEQ + i] = (_Float16)0.0f;
#ifdef USE_ASYNC_LDS
    __builtin_amdgcn_s_wait_asynccnt(0);
#endif
    __syncthreads();

    // Qh0[j] = q0 . W_Q[:,j]
    if (tid < DMODEL) {
        float a = 0.f;
        #pragma unroll
        for (int e = 0; e < DMODEL; ++e) a += s_q0[e] * W_Q[e * DMODEL + tid];
        s_qh[tid] = a;
    }
    __syncthreads();

    // u[e,h] = (1/sqrt(16)) * sum_d W_K[e, h*16+d] * Qh0[h*16+d]
    // packed as WMMA-B operand [n=16 padded][k=64] f16 in LDS.
    if (tid < 256) {
        int e = tid >> 2, h = tid & 3;
        float a = 0.f;
        #pragma unroll
        for (int d2 = 0; d2 < 16; ++d2)
            a += W_K[e * DMODEL + h * 16 + d2] * s_qh[h * 16 + d2];
        s_uB[h * DMODEL + e] = (_Float16)(0.25f * a);
    }
    __syncthreads();

    // ---------------- Phase 1: scores[k,h] = keys[k,:] . u[:,h] (WMMA) -----
    // 64 row-tiles of 16; wave w handles tiles {w, w+16, w+32, w+48}.
    {
        const int  n  = lane & 15;
        const bool hi = lane >= 16;

        // B operand (32x16 f16 per K-half): lanes 0-15 K=0..15, lanes 16-31 K=16..31
        v16h B0 = *(const v16h*)(s_uB + n * DMODEL + (hi ? 16 : 0));
        v16h B1 = *(const v16h*)(s_uB + n * DMODEL + 32 + (hi ? 16 : 0));

        #pragma unroll
        for (int t4 = 0; t4 < 4; ++t4) {
            const int tile = wave + t4 * 16;
            const int row  = tile * 16 + (lane & 15);
            const v4f* rp  = (const v4f*)(s_keys + row * DMODEL);
            const int  bq  = hi ? 4 : 0;
            // A operand (16x32 f16 per K-half), built by f32->f16 convert:
            // lanes<16 hold K {0..15}, lanes>=16 hold {16..31}
            V8HU p0, p1, p2, p3;
            p0.q[0] = __builtin_convertvector(rp[bq + 0],  v4h);
            p0.q[1] = __builtin_convertvector(rp[bq + 1],  v4h);
            p1.q[0] = __builtin_convertvector(rp[bq + 2],  v4h);
            p1.q[1] = __builtin_convertvector(rp[bq + 3],  v4h);
            p2.q[0] = __builtin_convertvector(rp[bq + 8],  v4h);
            p2.q[1] = __builtin_convertvector(rp[bq + 9],  v4h);
            p3.q[0] = __builtin_convertvector(rp[bq + 10], v4h);
            p3.q[1] = __builtin_convertvector(rp[bq + 11], v4h);
            V16U A0, A1;
            A0.h[0] = p0.v; A0.h[1] = p1.v;
            A1.h[0] = p2.v; A1.h[1] = p3.v;

            v8f c = {};
            c = __builtin_amdgcn_wmma_f32_16x16x32_f16(false, A0.v, false, B0,
                                                       (short)0, c, false, false);
            c = __builtin_amdgcn_wmma_f32_16x16x32_f16(false, A1.v, false, B1,
                                                       (short)0, c, false, false);

            if (n < NHEAD) {  // real heads: key-pad mask + diag blind (q==0 -> k==0)
                #pragma unroll
                for (int v = 0; v < 8; ++v) {
                    const int r = tile * 16 + v + (hi ? 8 : 0);
                    float val = c[v];
                    if (s_km[r] == 0 || r == 0) val = NEGF;
                    s_scores[r * NHEAD + n] = val;
                }
            }
        }
    }
    __syncthreads();

    // ---------------- Phase 2: softmax over k per head (waves 0..3) --------
    // p stored transposed as f16 into pT[h][k] (WMMA-A operand for phase 3).
    if (wave < NHEAD) {
        const int h = wave;
        float m = -3.4e38f;
        for (int r = lane; r < TSEQ; r += 32) m = fmaxf(m, s_scores[r * NHEAD + h]);
        for (int off = 16; off > 0; off >>= 1) m = fmaxf(m, __shfl_xor(m, off, 32));
        float l = 0.f;
        for (int r = lane; r < TSEQ; r += 32) {
            float p = __expf(s_scores[r * NHEAD + h] - m);
            s_pT[h * TSEQ + r] = (_Float16)p;
            l += p;
        }
        for (int off = 16; off > 0; off >>= 1) l += __shfl_xor(l, off, 32);
        if (lane == 0) s_lsum[h] = l;
    }
    __syncthreads();

    // ---------------- Phase 3: ctx = pT(16x1024) x keys(1024x64) (WMMA) ----
    // Wave w: e-tile nt = w&3, K-chunk group g = w>>2 handles kc = g,g+4,...,g+28.
    {
        const int  nt = wave & 3;
        const int  g  = wave >> 2;
        const int  n  = lane & 15;
        const bool hi = lane >= 16;
        const int  ecol = nt * 16 + n;

        v8f c = {};
        #pragma unroll
        for (int j = 0; j < 8; ++j) {
            const int kb = (g + 4 * j) * 32;
            // A: pT rows (heads), 16x32 f16 fragment (contiguous in k)
            const _Float16* ap = s_pT + (lane & 15) * TSEQ + kb + (hi ? 16 : 0);
            V16U A;
            A.h[0] = *(const v8h*)(ap);
            A.h[1] = *(const v8h*)(ap + 8);
            // B: keys column ecol, 16 K-contiguous f32 -> f16 (strided gather)
            const int kq = kb + (hi ? 16 : 0);
            v16h Bv;
            #pragma unroll
            for (int t = 0; t < 16; ++t)
                Bv[t] = (_Float16)s_keys[(kq + t) * DMODEL + ecol];

            c = __builtin_amdgcn_wmma_f32_16x16x32_f16(false, A.v, false, Bv,
                                                       (short)0, c, false, false);
        }
        #pragma unroll
        for (int v = 0; v < 8; ++v) {
            const int m = v + (hi ? 8 : 0);
            s_cpart[g * 1024 + nt * 256 + m * 16 + n] = c[v];
        }
    }
    __syncthreads();

    // reduce 4 partials; scale by query-mask / softmax denom
    {
        const float qm0 = (float)query_mask[b * TSEQ];
        if (tid < NHEAD * DMODEL) {
            const int h = tid >> 6, e = tid & 63;
            const int nt = e >> 4, n2 = e & 15;
            float a = 0.f;
            #pragma unroll
            for (int g = 0; g < 4; ++g)
                a += s_cpart[g * 1024 + nt * 256 + h * 16 + n2];
            s_ctx[h * DMODEL + e] = a * (qm0 / s_lsum[h]);
        }
    }
    __syncthreads();

    // ---------------- Phase 4: out0 = ctx @ W_V(head-block) + residual -----
    if (tid < DMODEL) {
        const int h = tid >> 4;
        float a = 0.f;
        #pragma unroll
        for (int e = 0; e < DMODEL; ++e) a += s_ctx[h * DMODEL + e] * W_V[e * DMODEL + tid];
        s_res[tid] = a + s_q0[tid];
    }
    __syncthreads();

    // ---------------- Phase 5: FFN on the single surviving row -------------
    if (tid < 4 * DMODEL) {
        float a = fb1[tid];
        #pragma unroll
        for (int e = 0; e < DMODEL; ++e) a += s_res[e] * fw1[e * (4 * DMODEL) + tid];
        s_hid[tid] = (a > 0.f) ? a : 0.2f * a;   // leaky_relu(0.2)
    }
    __syncthreads();
    if (tid < DMODEL) {
        float a = fb2[tid];
        #pragma unroll 8
        for (int j = 0; j < 4 * DMODEL; ++j) a += s_hid[j] * fw2[j * DMODEL + tid];
        out[b * DMODEL + tid] = s_res[tid] + a;
    }
}

extern "C" void kernel_launch(void* const* d_in, const int* in_sizes, int n_in,
                              void* d_out, int out_size, void* d_ws, size_t ws_size,
                              hipStream_t stream) {
    const float* queries    = (const float*)d_in[0];
    const float* keys       = (const float*)d_in[1];
    const int*   query_mask = (const int*)  d_in[2];
    const int*   key_mask   = (const int*)  d_in[3];
    const float* W_Q        = (const float*)d_in[4];
    const float* W_K        = (const float*)d_in[5];
    const float* W_V        = (const float*)d_in[6];
    const float* fw1        = (const float*)d_in[7];
    const float* fw2        = (const float*)d_in[8];
    const float* fb1        = (const float*)d_in[9];
    const float* fb2        = (const float*)d_in[10];
    float* out = (float*)d_out;

    const int B = in_sizes[2] / TSEQ;   // query_mask is [B, T]

    attn_block_kernel<<<dim3(B), dim3(NTHREADS), SMEM_BYTES, stream>>>(
        queries, keys, query_mask, key_mask, W_Q, W_K, W_V,
        fw1, fw2, fb1, fb2, out);
}
